// GAT_Model_88416196756296
// MI455X (gfx1250) — matrix operand order repaired and checked by reading, not compiled, not run on hardware
//
#include <hip/hip_runtime.h>

// Problem constants (match reference)
#define B_    8
#define N_    512
#define T_    128
#define DIN_  128
#define DOUT_ 128
#define H_    8
#define HD_   16
#define CH    128                 // rows (neighbors) per chunk
#define NCHUNK (N_ / CH)          // 4
#define ST_STRIDE 138             // f32 elems per sStage row (138%64=10 -> conflict-free A gathers,
                                  //   realized by TDM LDS padding: 128 DW interval + 10 DW pad)
#define HP_STRIDE 136             // bf16 elems per sHP row

typedef __attribute__((ext_vector_type(16))) __bf16 v16bf;
typedef __attribute__((ext_vector_type(8)))  float  v8f;
typedef __attribute__((ext_vector_type(4)))  unsigned int u32x4;
typedef __attribute__((ext_vector_type(8)))  int          i32x8;
typedef __attribute__((ext_vector_type(4)))  int          i32x4;

union BF16Vec { v16bf v; unsigned int u[8]; };

__device__ __forceinline__ unsigned int f2bf_u(float f) {
  // round-to-nearest-even f32 -> bf16 bits (used only for one-time W1 staging)
  unsigned int x = __float_as_uint(f);
  return (x + 0x7FFFu + ((x >> 16) & 1u)) >> 16;
}
__device__ __forceinline__ unsigned short f2bf_trunc(float f) {
  // truncating bf16: top 16 bits of the f32 register -> fuses to ds_store_b16_d16_hi
  return (unsigned short)(__float_as_uint(f) >> 16);
}
__device__ __forceinline__ unsigned int pack_bf16_trunc(float lo, float hi) {
  // one v_perm_b32: {hi[31:24],hi[23:16],lo[31:24],lo[23:16]}
  return __builtin_amdgcn_perm(__float_as_uint(hi), __float_as_uint(lo), 0x07060302u);
}
__device__ __forceinline__ float bf2f(unsigned short u) {
  return __uint_as_float(((unsigned int)u) << 16);
}

__global__ __launch_bounds__(256, 1)
void gat_fused_kernel(const float* __restrict__ h,
                      const float* __restrict__ adj,
                      const float* __restrict__ mask,
                      const float* __restrict__ W1,
                      const float* __restrict__ b1,
                      const float* __restrict__ Wa,
                      const float* __restrict__ ba,
                      const float* __restrict__ W2,
                      const float* __restrict__ b2,
                      float* __restrict__ out)
{
  // ~146 KB static LDS
  __shared__ __align__(16) unsigned int   sW1sw[8 * 4 * 256];    // 32 KB swizzled bf16 B-operands
  __shared__ __align__(16) float          sStage[CH * ST_STRIDE];// 69 KB f32 h-chunk (TDM destination)
  __shared__ __align__(16) unsigned short sHPb[CH * HP_STRIDE];  // 34 KB bf16 hp chunk
  __shared__ float sScore[H_ * CH];                              // 4 KB
  __shared__ float sE[H_ * CH];                                  // 4 KB (exp(s-M)*a; reused as agg at end)
  __shared__ float sAF[CH];
  __shared__ float sMz[CH];
  __shared__ float sTgt[DOUT_];
  __shared__ float sB1[DOUT_];
  __shared__ float sWsrc[HD_];
  __shared__ float sWdst[HD_];
  __shared__ float sS0[H_];

  const int tid  = threadIdx.x;
  const int lane = tid & 31;
  const int wave = tid >> 5;          // 8 waves; wave id == head id
  const int bid  = blockIdx.x;        // one workgroup per (b, t)
  const int b    = bid / T_;
  const int t    = bid - b * T_;

  // ---- TDM descriptor pieces that are chunk-invariant ----
  // Group1: data_size=4B (code 2), pad_enable, pad_interval code 6 (128 DW), pad_amount code 9 (10 DW),
  //         tensor_dim0 = DIN (128), tensor_dim1 = N (512), tile 128x128, dim0-stride = T*DIN = 16384.
  i32x8 g1;
  g1[0] = (int)((2u << 16) | (1u << 20) | (6u << 22) | (9u << 25));
  g1[1] = (int)(((unsigned)DIN_ & 0xFFFFu) << 16);   // tensor_dim0 low16 at bits 63:48
  g1[2] = (int)(((unsigned)N_  & 0xFFFFu) << 16);    // tensor_dim1 low16 at bits 95:80
  g1[3] = (int)(((unsigned)CH  & 0xFFFFu) << 16);    // tile_dim0 at bits 127:112
  g1[4] = (int)((unsigned)CH);                       // tile_dim1 at bits 143:128
  g1[5] = (int)(T_ * DIN_);                          // tensor_dim0_stride low32
  g1[6] = 0;
  g1[7] = 0;
  const i32x4 gz4 = {0, 0, 0, 0};
  const i32x8 gz8 = {0, 0, 0, 0, 0, 0, 0, 0};
  const unsigned int lds_stage = (unsigned int)(unsigned long long)(uintptr_t)&sStage[0];
  const unsigned long long hbase =
      (unsigned long long)(uintptr_t)h +
      ((unsigned long long)((size_t)b * N_) * T_ + (unsigned long long)t) * DIN_ * 4ull;
  const unsigned long long row_pitch_bytes = (unsigned long long)T_ * DIN_ * 4ull; // per n

  // ---------------- Phase 0: stage W1 (pre-swizzled bf16 B operands) + small vectors ----
  for (int idx = tid; idx < 8 * 4 * 256; idx += 256) {
    const int te = idx >> 10;
    const int ks = (idx >> 8) & 3;
    const int v  = (idx >> 5) & 7;
    const int ln = idx & 31;
    const int e  = te * 16 + (ln & 15);
    const int k  = ks * 32 + ((v < 4) ? 0 : 16) + (v & 3) * 2 + (ln >> 4) * 8;
    const unsigned int lo = f2bf_u(W1[k * DOUT_ + e]);
    const unsigned int hi = f2bf_u(W1[(k + 1) * DOUT_ + e]);
    sW1sw[idx] = lo | (hi << 16);
  }
  if (tid < DOUT_)                 sB1[tid]         = b1[tid];
  if (tid < HD_)                   sWsrc[tid]       = Wa[tid];
  else if (tid < 2 * HD_)          sWdst[tid - HD_] = Wa[tid];
  const float baval = ba[0];

  // ---- prologue: wave 0 launches TDM for chunk 0 ----
  if (wave == 0) {
    const unsigned long long ga = hbase;  // chunk 0 tile start
    u32x4 g0;
    g0[0] = 1u;                                  // count=1, user mode
    g0[1] = lds_stage;                           // lds_addr
    g0[2] = (unsigned int)(ga & 0xFFFFFFFFull);  // global_addr[31:0]
    g0[3] = (unsigned int)(ga >> 32) | (2u << 30); // global_addr[56:32] | type=2
    __builtin_amdgcn_tensor_load_to_lds(g0, g1, gz4, gz4, gz8, 0);
  }

  // per-lane online-softmax state (wave == head; lane owns d = lane&15, n-half = lane>>4)
  float rM = -INFINITY, rZ = 0.f, rN = 0.f;
  const int d_own = lane & 15;
  const int half  = lane >> 4;

  for (int chunk = 0; chunk < NCHUNK; ++chunk) {
    const int n0 = chunk * CH;

    if (wave == 0) __builtin_amdgcn_s_wait_tensorcnt(0);  // chunk's tile landed in sStage
    __syncthreads();  // publish sStage; also closes prior chunk's attention reads

    // ---- A-operand fetch: f32 sStage -> bf16 registers (once per chunk, reused by 8 N-tiles) ----
    const int tm   = wave;
    const int mrow = lane & 15;
    BF16Vec aop[4];
    #pragma unroll
    for (int ks = 0; ks < 4; ++ks) {
      #pragma unroll
      for (int v = 0; v < 8; ++v) {
        // 16-bit A layout: VGPR v holds K = ks*32 + (v<4?0:16) + (v%4)*2 + half*8 (+1)
        const int kk = ks * 32 + ((v < 4) ? 0 : 16) + (v & 3) * 2 + half * 8;
        const float f0 = sStage[(tm * 16 + mrow) * ST_STRIDE + kk];
        const float f1 = sStage[(tm * 16 + mrow) * ST_STRIDE + kk + 1];
        aop[ks].u[v] = pack_bf16_trunc(f0, f1);   // single v_perm_b32
      }
    }
    // ---- mask / adj reweight factor for this chunk ----
    if (tid < CH) {
      const int n = n0 + tid;
      const float m = mask[(size_t)(b * N_ + n) * T_ + t];
      float a = adj[(size_t)(b * T_ + t) * N_ + n];
      a = (a == 0.f) ? 1e9f : a;
      a = (m > 0.f) ? (1.f / a) : a;
      sAF[tid] = a;
      sMz[tid] = m;
    }
    __syncthreads();  // sStage consumed -> safe to refill

    // ---- overlap: wave 0 launches TDM for chunk+1 while everyone runs GEMM/attention ----
    if (wave == 0 && chunk + 1 < NCHUNK) {
      const unsigned long long ga = hbase + (unsigned long long)(n0 + CH) * row_pitch_bytes;
      u32x4 g0;
      g0[0] = 1u;
      g0[1] = lds_stage;
      g0[2] = (unsigned int)(ga & 0xFFFFFFFFull);
      g0[3] = (unsigned int)(ga >> 32) | (2u << 30);
      __builtin_amdgcn_tensor_load_to_lds(g0, g1, gz4, gz4, gz8, 0);
    }

    // ---- GEMM1 chunk: (128x128 bf16) @ W1(128x128 bf16) -> f32 acc -> bf16 hp ----
    #pragma unroll
    for (int te = 0; te < 8; ++te) {
      v8f acc = {0.f, 0.f, 0.f, 0.f, 0.f, 0.f, 0.f, 0.f};
      #pragma unroll
      for (int ks = 0; ks < 4; ++ks) {
        BF16Vec bop;
        #pragma unroll
        for (int v = 0; v < 8; ++v)
          bop.u[v] = sW1sw[(te * 4 + ks) * 256 + v * 32 + lane];
        acc = __builtin_amdgcn_wmma_f32_16x16x32_bf16(
                false, aop[ks].v, false, bop.v, (short)0, acc, false, false);
      }
      const int col = te * 16 + mrow;
      const float bb = sB1[col];
      #pragma unroll
      for (int i = 0; i < 8; ++i) {   // C/D layout: VGPR i -> M = i + 8*half, N = lane%16
        const int r = tm * 16 + i + half * 8;
        sHPb[r * HP_STRIDE + col] = f2bf_trunc(acc[i] + bb);  // ds_store_b16_d16_hi
      }
    }
    __syncthreads();

    // ---- chunk 0 only: source-node score term s0[h], and target_h row ----
    if (chunk == 0) {
      if (tid < H_) {
        float s = 0.f;
        #pragma unroll
        for (int j = 0; j < HD_; ++j) s += bf2f(sHPb[tid * HD_ + j]) * sWsrc[j];
        sS0[tid] = s;
      } else if (tid >= 128 && tid < 128 + DOUT_) {
        sTgt[tid - 128] = bf2f(sHPb[tid - 128]);
      }
      __syncthreads();
    }

    // ---- per-(head, node) raw scores ----
    for (int p = tid; p < H_ * CH; p += 256) {
      const int hh = p >> 7;
      const int n  = p & (CH - 1);
      float dot = 0.f;
      #pragma unroll
      for (int j = 0; j < HD_; ++j)
        dot += bf2f(sHPb[n * HP_STRIDE + hh * HD_ + j]) * sWdst[j];
      float sc = sS0[hh] + dot + baval;
      if (sMz[n] == 0.f) sc = -1e9f;
      sScore[hh * CH + n] = sc;
    }
    __syncthreads();

    // ---- online softmax + weighted aggregation (wave w handles head w) ----
    {
      const int hh = wave;
      float scv[4];
      float lmax = -INFINITY;
      #pragma unroll
      for (int i = 0; i < 4; ++i) {
        scv[i] = sScore[hh * CH + lane + 32 * i];
        lmax = fmaxf(lmax, scv[i]);
      }
      #pragma unroll
      for (int off = 16; off >= 1; off >>= 1)
        lmax = fmaxf(lmax, __shfl_xor(lmax, off, 32));
      const float newM  = fmaxf(rM, lmax);
      const float scale = __expf(rM - newM);           // rM = -inf first time -> 0
      float zc = 0.f;
      #pragma unroll
      for (int i = 0; i < 4; ++i) {
        const int n = lane + 32 * i;
        const float e = __expf(scv[i] - newM);
        zc += e;
        sE[hh * CH + n] = e * sAF[n];                  // numerator weight includes adj factor
      }
      #pragma unroll
      for (int off = 16; off >= 1; off >>= 1)
        zc += __shfl_xor(zc, off, 32);
      rZ = rZ * scale + zc;                            // Z excludes adj factor (softmax denom)
      rM = newM;
      rN *= scale;
      const int nbase = half * 64;                     // two lane-halves split the n range
      for (int n = nbase; n < nbase + 64; ++n)
        rN += sE[hh * CH + n] * bf2f(sHPb[n * HP_STRIDE + hh * HD_ + d_own]);
    }
    __syncthreads();
  }

  // ---- finalize aggregation: agg[h*16+d] = N / Z  (reuse sE[0..127]) ----
  {
    const float tot = rN + __shfl_xor(rN, 16, 32);     // combine the two n-halves
    if (lane < 16) sE[wave * HD_ + lane] = tot / rZ;
  }
  __syncthreads();

  // ---- final row GEMM: out = [agg | target_h] @ W2 + b2 ----
  if (tid < DOUT_) {
    const int e = tid;
    float acc = b2[e];
    for (int k = 0; k < DOUT_; ++k) acc += sE[k]   * W2[k * DOUT_ + e];
    for (int k = 0; k < DOUT_; ++k) acc += sTgt[k] * W2[(DOUT_ + k) * DOUT_ + e];
    out[(size_t)(b * T_ + t) * DOUT_ + e] = acc;
  }
}

extern "C" void kernel_launch(void* const* d_in, const int* in_sizes, int n_in,
                              void* d_out, int out_size, void* d_ws, size_t ws_size,
                              hipStream_t stream) {
  (void)in_sizes; (void)n_in; (void)d_ws; (void)ws_size; (void)out_size;
  const float* h    = (const float*)d_in[0];
  const float* adj  = (const float*)d_in[1];
  const float* mask = (const float*)d_in[2];
  const float* W1   = (const float*)d_in[3];
  const float* b1   = (const float*)d_in[4];
  const float* Wa   = (const float*)d_in[5];
  const float* ba   = (const float*)d_in[6];
  const float* W2   = (const float*)d_in[7];
  const float* b2   = (const float*)d_in[8];
  float* out = (float*)d_out;

  gat_fused_kernel<<<dim3(B_ * T_), dim3(256), 0, stream>>>(
      h, adj, mask, W1, b1, Wa, ba, W2, b2, out);
}
